// GaussianMixer_83769042141845
// MI455X (gfx1250) — matrix operand order repaired and checked by reading, not compiled, not run on hardware
//
#include <hip/hip_runtime.h>
#include <hip/hip_bf16.h>

typedef _Float16 f16;
typedef __attribute__((ext_vector_type(2)))  _Float16 h2;
typedef __attribute__((ext_vector_type(16))) _Float16 v16h;
typedef __attribute__((ext_vector_type(8)))  float    v8f;

#define NGS   2048
#define IMH   128
#define IMW   128
#define HW    (IMH*IMW)
#define EPSN  1e-5f

#define WMMA_F16(a,b,c) __builtin_amdgcn_wmma_f32_16x16x32_f16(false,(a),false,(b),(short)0,(c),false,false)

// ---------------- workspace layout (float offsets) ----------------
#define O_PARAMS  0u          // 2048*8
#define O_KEYS    16384u      // 2048
#define O_IDX     18432u      // 2048 int
#define O_SIDX    20480u      // 2048 int
#define O_FEATS   22528u      // 2048*16
#define O_SPARAMS 55296u      // 2048*6
#define O_SFEATS  67584u      // 2048*16 f16, stored in B-fragment order
#define O_IMG     83968u      // 16*128*128
#define O_R0      346112u     // 32*64*64
#define O_R1      477184u     // 64*32*32
#define O_R2      542720u     // 128*16*16
#define O_R3      575488u     // 256*8*8
#define O_T0      591872u     // 128*16*16
#define O_T1      624640u     // 64*32*32
#define O_T2      690176u     // 32*64*64
#define O_T3      821248u     // 16*128*128
#define O_FIN     1083392u    // 16*128*128
#define O_STATS   1345536u    // 20 floats

// ---------------- fragment-order swizzles (CDNA5 wave32 WMMA layouts) ----------------
// A 16x32 f16 fragment: lane l holds row M=l&15; elem j -> K = 16*(j>>3) + 8*(l>>4) + (j&7)
// inverse: element (m,k) lives at lane = m + 16*((k>>3)&1), j = 8*(k>>4) + (k&7)
__device__ __forceinline__ int aSwz(int m, int k) {
  int lane = (m & 15) + (((k >> 3) & 1) << 4);
  int j = ((k >> 4) << 3) + (k & 7);
  return lane * 16 + j;
}
// B 32x16 f16 fragment: lane l holds col N=l&15; elem j -> K = 16*(j>>3) + 8*(l>>4) + (j&7)
// inverse: element (k,n) lives at lane = n + 16*((k>>3)&1), j = 8*(k>>4) + (k&7)
__device__ __forceinline__ int bSwz(int k, int n) {
  int lane = (n & 15) + (((k >> 3) & 1) << 4);
  int j = ((k >> 4) << 3) + (k & 7);
  return lane * 16 + j;
}
// a tile stored fragment-order is 512 halves; per-lane data is 32 contiguous bytes
__device__ __forceinline__ v16h fragLoad(const f16* tileBase) {
  return *(const v16h*)(tileBase + (threadIdx.x & 31) * 16);
}
// C/D 16x16 f32: lane l -> col N=l&15; vgpr r -> row M = r + 8*(l>>4)
__device__ __forceinline__ void stD16x16(const v8f& d, float* base, int ld, float bn) {
  int l = threadIdx.x & 31;
  int n = l & 15, mh = (l >> 4) * 8;
#pragma unroll
  for (int r = 0; r < 8; ++r) base[(mh + r) * ld + n] = d[r] + bn;
}

// per-row LayerNorm (64 wide) + ReLU: f32 row-major LDS -> f16 A-fragment-order LDS
__device__ __forceinline__ void ln_relu_swz(const float* H, f16* X,
                                            const float* g, const float* q, int tid) {
  if (tid < 128) {
    const float* h = H + tid * 64;
    float mu = 0.f;
#pragma unroll
    for (int c = 0; c < 64; ++c) mu += h[c];
    mu *= (1.f / 64.f);
    float var = 0.f;
#pragma unroll
    for (int c = 0; c < 64; ++c) { float d = h[c] - mu; var += d * d; }
    var *= (1.f / 64.f);
    float rs = rsqrtf(var + EPSN);
    int tile = tid >> 4, m = tid & 15;
#pragma unroll
    for (int c = 0; c < 64; ++c) {
      float y = (h[c] - mu) * rs * g[c] + q[c];
      X[((tile << 1) + (c >> 5)) * 512 + aSwz(m, c & 31)] = (f16)fmaxf(y, 0.f);
    }
  }
}

// ---------------- gaussian parameter prep ----------------
__global__ void k_prep(const float* __restrict__ pos, const float* __restrict__ ls,
                       const float* __restrict__ rot, const float* __restrict__ al,
                       const float* __restrict__ zd,
                       float* __restrict__ params, float* __restrict__ keys,
                       int* __restrict__ idx) {
  int i = blockIdx.x * blockDim.x + threadIdx.x;
  if (i >= NGS) return;
  float sx2 = __expf(2.f * ls[2 * i + 0]);
  float sy2 = __expf(2.f * ls[2 * i + 1]);
  float c = __cosf(rot[i]), s = __sinf(rot[i]);
  float a = c * c * sx2 + s * s * sy2;
  float b = c * s * (sx2 - sy2);
  float d = s * s * sx2 + c * c * sy2;
  float det = a * d - b * b;
  float ia = d / det, ib = -b / det, ic = a / det;
  float px = (pos[2 * i + 0] + 1.f) * 0.5f * (float)IMW;
  float py = (pos[2 * i + 1] + 1.f) * 0.5f * (float)IMH;
  float alpha = 1.f / (1.f + __expf(-al[i]));
  float dep = fminf(fmaxf(zd[i], 0.f), 1.f);
  float* p = params + i * 8;
  p[0] = px; p[1] = py; p[2] = ia; p[3] = ib; p[4] = ic; p[5] = alpha; p[6] = dep; p[7] = 0.f;
  keys[i] = dep;
  idx[i] = i;
}

// ---------------- single-block bitonic argsort of 2048 depths ----------------
__global__ void k_sort(const float* __restrict__ keys, const int* __restrict__ idx,
                       int* __restrict__ sidx) {
  __shared__ float kk[NGS];
  __shared__ int   vv[NGS];
  int t = threadIdx.x;
  for (int i = t; i < NGS; i += 1024) { kk[i] = keys[i]; vv[i] = idx[i]; }
  __syncthreads();
  for (int k = 2; k <= NGS; k <<= 1) {
    for (int j = k >> 1; j > 0; j >>= 1) {
      int lo = t & (j - 1);
      int i  = (t << 1) - lo;
      int p  = i + j;
      bool up = ((i & k) == 0);
      float ki = kk[i], kp = kk[p];
      bool sw = up ? (ki > kp) : (ki < kp);
      if (sw) {
        kk[i] = kp; kk[p] = ki;
        int tv = vv[i]; vv[i] = vv[p]; vv[p] = tv;
      }
      __syncthreads();
    }
  }
  for (int i = t; i < NGS; i += 1024) sidx[i] = vv[i];
}

// ---------------- init MLP (32->64 LN relu, 64->64 LN relu, 64->16) via WMMA ----------------
__global__ void k_mlp(const float* __restrict__ x,
                      const float* w1, const float* b1, const float* g1, const float* q1,
                      const float* w2, const float* b2, const float* g2, const float* q2,
                      const float* w3, const float* b3,
                      float* __restrict__ feats) {
  __shared__ alignas(32) f16   Xa[128 * 64];   // A tiles, fragment order
  __shared__ alignas(32) f16   Wl[64 * 64];    // B tiles, fragment order
  __shared__ alignas(32) float Hb[128 * 64];   // GEMM out, row-major for LN
  __shared__ float bp[64], gp[64], qp[64];
  int tid = threadIdx.x;
  int wave = tid >> 5;
  int rowBase = blockIdx.x * 128;

  // stage input (K=32, one chunk): A tiles [8][512]
  for (int i = tid; i < 128 * 32; i += 256) {
    int row = i >> 5, k = i & 31;
    Xa[(row >> 4) * 512 + aSwz(row & 15, k)] = (f16)x[rowBase * 32 + i];
  }
  // W1 [32][64] -> B tiles [nt=4][512]
  for (int i = tid; i < 32 * 64; i += 256) {
    int k = i >> 6, n = i & 63;
    Wl[(n >> 4) * 512 + bSwz(k, n & 15)] = (f16)w1[i];
  }
  if (tid < 64) { bp[tid] = b1[tid]; gp[tid] = g1[tid]; qp[tid] = q1[tid]; }
  __syncthreads();

  // GEMM1: [128x32] @ [32x64]
  for (int nt = 0; nt < 4; ++nt) {
    v8f acc = {};
    v16h a = fragLoad(Xa + wave * 512);
    v16h b = fragLoad(Wl + nt * 512);
    acc = WMMA_F16(a, b, acc);
    stD16x16(acc, Hb + wave * 16 * 64 + nt * 16, 64, bp[nt * 16 + (tid & 15)]);
  }
  __syncthreads();
  ln_relu_swz(Hb, Xa, gp, qp, tid);
  __syncthreads();
  // W2 [64][64] -> B tiles [nt*2+kc][512]
  for (int i = tid; i < 64 * 64; i += 256) {
    int k = i >> 6, n = i & 63;
    Wl[(((n >> 4) << 1) + (k >> 5)) * 512 + bSwz(k & 31, n & 15)] = (f16)w2[i];
  }
  if (tid < 64) { bp[tid] = b2[tid]; gp[tid] = g2[tid]; qp[tid] = q2[tid]; }
  __syncthreads();

  // GEMM2: [128x64] @ [64x64]
  for (int nt = 0; nt < 4; ++nt) {
    v8f acc = {};
    for (int kc = 0; kc < 2; ++kc) {
      v16h a = fragLoad(Xa + ((wave << 1) + kc) * 512);
      v16h b = fragLoad(Wl + ((nt << 1) + kc) * 512);
      acc = WMMA_F16(a, b, acc);
    }
    stD16x16(acc, Hb + wave * 16 * 64 + nt * 16, 64, bp[nt * 16 + (tid & 15)]);
  }
  __syncthreads();
  ln_relu_swz(Hb, Xa, gp, qp, tid);
  __syncthreads();
  // W3 [64][16] -> B tiles [kc=2][512]
  for (int i = tid; i < 64 * 16; i += 256) {
    int k = i >> 4, n = i & 15;
    Wl[(k >> 5) * 512 + bSwz(k & 31, n)] = (f16)w3[i];
  }
  if (tid < 16) bp[tid] = b3[tid];
  __syncthreads();

  // GEMM3: [128x64] @ [64x16] -> global feats
  {
    v8f acc = {};
    for (int kc = 0; kc < 2; ++kc) {
      v16h a = fragLoad(Xa + ((wave << 1) + kc) * 512);
      v16h b = fragLoad(Wl + kc * 512);
      acc = WMMA_F16(a, b, acc);
    }
    int l = tid & 31;
    int n = l & 15, mh = (l >> 4) * 8;
    float bv = bp[n];
#pragma unroll
    for (int r = 0; r < 8; ++r)
      feats[(rowBase + wave * 16 + mh + r) * 16 + n] = acc[r] + bv;
  }
}

// ---------------- gather sorted params + f16 feats in B-fragment order ----------------
__global__ void k_pack(const int* __restrict__ sidx, const float* __restrict__ params,
                       const float* __restrict__ feats,
                       float* __restrict__ sp, f16* __restrict__ sf) {
  int i = blockIdx.x * blockDim.x + threadIdx.x;
  if (i >= NGS) return;
  int src = sidx[i];
#pragma unroll
  for (int j = 0; j < 6; ++j) sp[i * 6 + j] = params[src * 8 + j];
  int t = i >> 5, kk = i & 31;
#pragma unroll
  for (int n = 0; n < 16; ++n)
    sf[t * 512 + bSwz(kk, n)] = (f16)feats[src * 16 + n];
}

// ---------------- rasterizer: transmittance scan + WMMA splat accumulate ----------------
__global__ void k_raster(const float* __restrict__ sp, const f16* __restrict__ sf,
                         float* __restrict__ img) {
  __shared__ alignas(32) f16 F[NGS * 16];        // all sorted feats (B-frag order), 64 KB
  __shared__ alignas(32) f16 wbuf[8 * 2 * 512];  // per-wave A tiles, 16 KB
  int tid = threadIdx.x;
  {  // straight vector copy: global (already fragment order) -> LDS
    const int4* s4 = (const int4*)sf;
    int4* d4 = (int4*)F;
    for (int i = tid; i < NGS * 16 * (int)sizeof(f16) / 16; i += 256) d4[i] = s4[i];
  }
  __syncthreads();

  int wave = tid >> 5, l = tid & 31;
  int pbase = blockIdx.x * 256 + wave * 32;
  int p = pbase + l;
  float fx = (float)(p % IMW) + 0.5f;
  float fy = (float)(p / IMW) + 0.5f;
  float T = 1.f;
  v8f acc0 = {}, acc1 = {};
  f16* wb = wbuf + wave * 1024;
  int m = l & 15, atile = l >> 4;

  for (int tile = 0; tile < NGS / 32; ++tile) {
    const float* pp = sp + tile * 32 * 6;   // wave-uniform -> scalar loads
    for (int k2 = 0; k2 < 16; ++k2) {
      int k = k2 << 1;
      h2 wpair;
#pragma unroll
      for (int s = 0; s < 2; ++s) {
        const float* q = pp + (k + s) * 6;
        float dx = fx - q[0], dy = fy - q[1];
        float qv = q[2] * dx * dx + 2.f * q[3] * dx * dy + q[4] * dy * dy;
        float g  = __expf(-0.5f * qv);
        float a  = fminf(q[5] * g, 0.995f);
        wpair[s] = (f16)(a * T);
        T *= (1.f - a);
      }
      // even k -> even j: pair is one aligned 32-bit LDS store
      *(h2*)(wb + atile * 512 + aSwz(m, k)) = wpair;
    }
    // wave-private LDS: DS ops in-order per wave, no barrier needed
    v16h a0 = fragLoad(wb);          // pixels pbase+0..15
    v16h a1 = fragLoad(wb + 512);    // pixels pbase+16..31
    v16h b  = fragLoad(F + tile * 512);
    acc0 = WMMA_F16(a0, b, acc0);
    acc1 = WMMA_F16(a1, b, acc1);
  }
  int n = l & 15, mh = (l >> 4) * 8;
#pragma unroll
  for (int r = 0; r < 8; ++r) {
    img[n * HW + pbase + mh + r]      = acc0[r];   // CHW for the UNet
    img[n * HW + pbase + 16 + mh + r] = acc1[r];
  }
}

// ---------------- UNet pieces ----------------
__global__ void k_conv3(const float* __restrict__ in, const float* __restrict__ w,
                        const float* __restrict__ b, float* __restrict__ out,
                        int Ci, int Co, int Hi, int Wi, int stride) {
  int Ho = Hi / stride, Wo = Wi / stride;
  int total = Co * Ho * Wo;
  int i = blockIdx.x * 256 + threadIdx.x;
  if (i >= total) return;
  int ox = i % Wo, oy = (i / Wo) % Ho, oc = i / (Wo * Ho);
  float acc = b[oc];
  for (int ci = 0; ci < Ci; ++ci) {
    const float* ip = in + ci * Hi * Wi;
    const float* wp = w + ((oc * Ci + ci) * 9);
#pragma unroll
    for (int ky = 0; ky < 3; ++ky) {
      int iy = oy * stride - 1 + ky;
      if ((unsigned)iy >= (unsigned)Hi) continue;
#pragma unroll
      for (int kx = 0; kx < 3; ++kx) {
        int ix = ox * stride - 1 + kx;
        if ((unsigned)ix >= (unsigned)Wi) continue;
        acc += ip[iy * Wi + ix] * wp[ky * 3 + kx];
      }
    }
  }
  out[i] = acc;
}

__global__ void k_convT(const float* __restrict__ in, const float* __restrict__ w,
                        const float* __restrict__ b, float* __restrict__ out,
                        int Ci, int Co, int Hi, int Wi) {
  int Ho = Hi * 2, Wo = Wi * 2;
  int total = Co * Ho * Wo;
  int i = blockIdx.x * 256 + threadIdx.x;
  if (i >= total) return;
  int ox = i % Wo, oy = (i / Wo) % Ho, oc = i / (Wo * Ho);
  int iy = oy >> 1, ix = ox >> 1, ky = oy & 1, kx = ox & 1;
  float acc = b[oc];
  for (int ci = 0; ci < Ci; ++ci)
    acc += in[(ci * Hi + iy) * Wi + ix] * w[((ci * Co + oc) * 2 + ky) * 2 + kx];
  out[i] = acc;
}

__global__ void k_gnstats(const float* __restrict__ x, int n, float* __restrict__ stats) {
  __shared__ float ss[1024], sq[1024];
  int t = threadIdx.x;
  float s = 0.f, q = 0.f;
  for (int i = t; i < n; i += 1024) { float v = x[i]; s += v; q += v * v; }
  ss[t] = s; sq[t] = q;
  __syncthreads();
  for (int o = 512; o > 0; o >>= 1) {
    if (t < o) { ss[t] += ss[t + o]; sq[t] += sq[t + o]; }
    __syncthreads();
  }
  if (t == 0) {
    float mu = ss[0] / (float)n;
    float var = sq[0] / (float)n - mu * mu;
    stats[0] = mu;
    stats[1] = rsqrtf(var + EPSN);
  }
}

__global__ void k_gnorm(float* __restrict__ x, const float* __restrict__ stats,
                        const float* __restrict__ skip, int n) {
  int i = blockIdx.x * 256 + threadIdx.x;
  if (i >= n) return;
  float y = fmaxf(0.f, (x[i] - stats[0]) * stats[1]);
  if (skip) y += skip[i];
  x[i] = y;
}

// ---------------- grid_sample + up_proj + final MLP + head via WMMA ----------------
__global__ void k_head(const float* __restrict__ pos, const float* __restrict__ fin,
                       const float* wup, const float* bup,
                       const float* w0, const float* b0, const float* g0, const float* q0,
                       const float* w1, const float* b1, const float* g1, const float* q1,
                       const float* wout, const float* bout,
                       float* __restrict__ out) {
  __shared__ alignas(32) f16   Xa[128 * 64];
  __shared__ alignas(32) f16   Wl[64 * 64];
  __shared__ alignas(32) float Hb[128 * 64];
  __shared__ float bp[64], gp[64], qp[64];
  int tid = threadIdx.x;
  int wave = tid >> 5;
  int rowBase = blockIdx.x * 128;

  // bilinear grid_sample (zeros padding, align_corners=False) -> A tiles (K padded to 32)
  if (tid < 128) {
    int g = rowBase + tid;
    float px = pos[2 * g + 0], py = pos[2 * g + 1];
    float ix = ((px + 1.f) * (float)IMW - 1.f) * 0.5f;
    float iy = ((py + 1.f) * (float)IMH - 1.f) * 0.5f;
    float x0f = floorf(ix), y0f = floorf(iy);
    float wx = ix - x0f, wy = iy - y0f;
    int x0 = (int)x0f, y0 = (int)y0f;
    int tile = tid >> 4, m = tid & 15;
#pragma unroll
    for (int c = 0; c < 16; ++c) {
      const float* ip = fin + c * HW;
      float v00 = 0.f, v01 = 0.f, v10 = 0.f, v11 = 0.f;
      if ((unsigned)x0 < (unsigned)IMW && (unsigned)y0 < (unsigned)IMH) v00 = ip[y0 * IMW + x0];
      if ((unsigned)(x0 + 1) < (unsigned)IMW && (unsigned)y0 < (unsigned)IMH) v01 = ip[y0 * IMW + x0 + 1];
      if ((unsigned)x0 < (unsigned)IMW && (unsigned)(y0 + 1) < (unsigned)IMH) v10 = ip[(y0 + 1) * IMW + x0];
      if ((unsigned)(x0 + 1) < (unsigned)IMW && (unsigned)(y0 + 1) < (unsigned)IMH) v11 = ip[(y0 + 1) * IMW + x0 + 1];
      float v = v00 * (1 - wx) * (1 - wy) + v01 * wx * (1 - wy) + v10 * (1 - wx) * wy + v11 * wx * wy;
      Xa[tile * 512 + aSwz(m, c)] = (f16)v;
    }
#pragma unroll
    for (int c = 16; c < 32; ++c) Xa[tile * 512 + aSwz(m, c)] = (f16)0.f;
  }
  // W_up [16][64] zero-padded to K=32, B tiles [nt=4][512]
  for (int i = tid; i < 32 * 64; i += 256) {
    int k = i >> 6, n = i & 63;
    f16 v = (k < 16) ? (f16)wup[k * 64 + n] : (f16)0.f;
    Wl[(n >> 4) * 512 + bSwz(k, n & 15)] = v;
  }
  if (tid < 64) bp[tid] = bup[tid];
  __syncthreads();

  // up_proj: [128x32(pad)] @ [32x64]
  for (int nt = 0; nt < 4; ++nt) {
    v8f acc = {};
    v16h a = fragLoad(Xa + wave * 512);
    v16h b = fragLoad(Wl + nt * 512);
    acc = WMMA_F16(a, b, acc);
    stD16x16(acc, Hb + wave * 16 * 64 + nt * 16, 64, bp[nt * 16 + (tid & 15)]);
  }
  __syncthreads();
  // plain linear (no LN): re-stage Hb as A tiles [tile*2+kc][512]
  for (int i = tid; i < 128 * 64; i += 256) {
    int row = i >> 6, c = i & 63;
    Xa[(((row >> 4) << 1) + (c >> 5)) * 512 + aSwz(row & 15, c & 31)] = (f16)Hb[i];
  }
  for (int i = tid; i < 64 * 64; i += 256) {
    int k = i >> 6, n = i & 63;
    Wl[(((n >> 4) << 1) + (k >> 5)) * 512 + bSwz(k & 31, n & 15)] = (f16)w0[i];
  }
  if (tid < 64) { bp[tid] = b0[tid]; gp[tid] = g0[tid]; qp[tid] = q0[tid]; }
  __syncthreads();

  // final_mlp[0]
  for (int nt = 0; nt < 4; ++nt) {
    v8f acc = {};
    for (int kc = 0; kc < 2; ++kc) {
      v16h a = fragLoad(Xa + ((wave << 1) + kc) * 512);
      v16h b = fragLoad(Wl + ((nt << 1) + kc) * 512);
      acc = WMMA_F16(a, b, acc);
    }
    stD16x16(acc, Hb + wave * 16 * 64 + nt * 16, 64, bp[nt * 16 + (tid & 15)]);
  }
  __syncthreads();
  ln_relu_swz(Hb, Xa, gp, qp, tid);
  __syncthreads();
  for (int i = tid; i < 64 * 64; i += 256) {
    int k = i >> 6, n = i & 63;
    Wl[(((n >> 4) << 1) + (k >> 5)) * 512 + bSwz(k & 31, n & 15)] = (f16)w1[i];
  }
  if (tid < 64) { bp[tid] = b1[tid]; gp[tid] = g1[tid]; qp[tid] = q1[tid]; }
  __syncthreads();

  // final_mlp[1]
  for (int nt = 0; nt < 4; ++nt) {
    v8f acc = {};
    for (int kc = 0; kc < 2; ++kc) {
      v16h a = fragLoad(Xa + ((wave << 1) + kc) * 512);
      v16h b = fragLoad(Wl + ((nt << 1) + kc) * 512);
      acc = WMMA_F16(a, b, acc);
    }
    stD16x16(acc, Hb + wave * 16 * 64 + nt * 16, 64, bp[nt * 16 + (tid & 15)]);
  }
  __syncthreads();
  ln_relu_swz(Hb, Xa, gp, qp, tid);
  __syncthreads();

  // final_out: 64 -> 3 (reads A-fragment-order activations)
  if (tid < 128) {
    int g = rowBase + tid;
    int tile = tid >> 4, m = tid & 15;
#pragma unroll
    for (int o = 0; o < 3; ++o) {
      float acc = bout[o];
      for (int c = 0; c < 64; ++c)
        acc += (float)Xa[((tile << 1) + (c >> 5)) * 512 + aSwz(m, c & 31)] * wout[c * 3 + o];
      out[g * 3 + o] = acc;
    }
  }
}

// ---------------- launch ----------------
extern "C" void kernel_launch(void* const* d_in, const int* in_sizes, int n_in,
                              void* d_out, int out_size, void* d_ws, size_t ws_size,
                              hipStream_t stream) {
  (void)in_sizes; (void)n_in; (void)out_size; (void)ws_size;
  const float* x   = (const float*)d_in[0];
  const float* pos = (const float*)d_in[1];
  const float* ls  = (const float*)d_in[2];
  const float* rot = (const float*)d_in[3];
  const float* al  = (const float*)d_in[4];
  const float* zd  = (const float*)d_in[5];
  const float* wi0 = (const float*)d_in[6];  const float* bi0 = (const float*)d_in[7];
  const float* gi0 = (const float*)d_in[8];  const float* qi0 = (const float*)d_in[9];
  const float* wi1 = (const float*)d_in[10]; const float* bi1 = (const float*)d_in[11];
  const float* gi1 = (const float*)d_in[12]; const float* qi1 = (const float*)d_in[13];
  const float* wdp = (const float*)d_in[14]; const float* bdp = (const float*)d_in[15];
  const float* wup = (const float*)d_in[16]; const float* bup = (const float*)d_in[17];
  const float* wd[4] = {(const float*)d_in[18],(const float*)d_in[20],(const float*)d_in[22],(const float*)d_in[24]};
  const float* bd[4] = {(const float*)d_in[19],(const float*)d_in[21],(const float*)d_in[23],(const float*)d_in[25]};
  const float* wu[4] = {(const float*)d_in[26],(const float*)d_in[28],(const float*)d_in[30],(const float*)d_in[32]};
  const float* bu[4] = {(const float*)d_in[27],(const float*)d_in[29],(const float*)d_in[31],(const float*)d_in[33]};
  const float* wf  = (const float*)d_in[34]; const float* bf  = (const float*)d_in[35];
  const float* wm0 = (const float*)d_in[36]; const float* bm0 = (const float*)d_in[37];
  const float* gm0 = (const float*)d_in[38]; const float* qm0 = (const float*)d_in[39];
  const float* wm1 = (const float*)d_in[40]; const float* bm1 = (const float*)d_in[41];
  const float* gm1 = (const float*)d_in[42]; const float* qm1 = (const float*)d_in[43];
  const float* wo  = (const float*)d_in[44]; const float* bo  = (const float*)d_in[45];

  float* ws    = (float*)d_ws;
  float* Pm    = ws + O_PARAMS;
  float* keys  = ws + O_KEYS;
  int*   idx   = (int*)(ws + O_IDX);
  int*   sidx  = (int*)(ws + O_SIDX);
  float* feats = ws + O_FEATS;
  float* sp    = ws + O_SPARAMS;
  f16*   sf    = (f16*)(ws + O_SFEATS);
  float* img   = ws + O_IMG;
  float* r0 = ws + O_R0; float* r1 = ws + O_R1; float* r2 = ws + O_R2; float* r3 = ws + O_R3;
  float* t0 = ws + O_T0; float* t1 = ws + O_T1; float* t2 = ws + O_T2; float* t3 = ws + O_T3;
  float* fin = ws + O_FIN;
  float* st  = ws + O_STATS;
  float* out = (float*)d_out;

  k_prep<<<8, 256, 0, stream>>>(pos, ls, rot, al, zd, Pm, keys, idx);
  k_sort<<<1, 1024, 0, stream>>>(keys, idx, sidx);
  k_mlp<<<16, 256, 0, stream>>>(x, wi0, bi0, gi0, qi0, wi1, bi1, gi1, qi1, wdp, bdp, feats);
  k_pack<<<8, 256, 0, stream>>>(sidx, Pm, feats, sp, sf);
  k_raster<<<64, 256, 0, stream>>>(sp, sf, img);

  // UNet down path: conv(s2) -> GN -> ReLU (in place)
  struct { const float* in; float* outb; int Ci, Co, Hi; } dn[4] = {
    { img, r0, 16, 32, 128 }, { r0, r1, 32, 64, 64 },
    { r1, r2, 64, 128, 32 }, { r2, r3, 128, 256, 16 } };
  for (int i = 0; i < 4; ++i) {
    int n = dn[i].Co * (dn[i].Hi / 2) * (dn[i].Hi / 2);
    k_conv3<<<(n + 255) / 256, 256, 0, stream>>>(dn[i].in, wd[i], bd[i], dn[i].outb,
                                                 dn[i].Ci, dn[i].Co, dn[i].Hi, dn[i].Hi, 2);
    k_gnstats<<<1, 1024, 0, stream>>>(dn[i].outb, n, st + 2 * i);
    k_gnorm<<<(n + 255) / 256, 256, 0, stream>>>(dn[i].outb, st + 2 * i, nullptr, n);
  }
  // UNet up path: convT -> GN -> ReLU -> + skip (skips: r2, r1, r0, img)
  struct { const float* in; float* outb; const float* skip; int Ci, Co, Hi; } up[4] = {
    { r3, t0, r2, 256, 128, 8 }, { t0, t1, r1, 128, 64, 16 },
    { t1, t2, r0, 64, 32, 32 }, { t2, t3, img, 32, 16, 64 } };
  for (int i = 0; i < 4; ++i) {
    int n = up[i].Co * (up[i].Hi * 2) * (up[i].Hi * 2);
    k_convT<<<(n + 255) / 256, 256, 0, stream>>>(up[i].in, wu[i], bu[i], up[i].outb,
                                                 up[i].Ci, up[i].Co, up[i].Hi, up[i].Hi);
    k_gnstats<<<1, 1024, 0, stream>>>(up[i].outb, n, st + 2 * (4 + i));
    k_gnorm<<<(n + 255) / 256, 256, 0, stream>>>(up[i].outb, st + 2 * (4 + i), up[i].skip, n);
  }
  // final conv(s1) -> GN -> ReLU
  {
    int n = 16 * 128 * 128;
    k_conv3<<<(n + 255) / 256, 256, 0, stream>>>(t3, wf, bf, fin, 16, 16, 128, 128, 1);
    k_gnstats<<<1, 1024, 0, stream>>>(fin, n, st + 16);
    k_gnorm<<<(n + 255) / 256, 256, 0, stream>>>(fin, st + 16, nullptr, n);
  }

  k_head<<<16, 256, 0, stream>>>(pos, fin, wup, bup, wm0, bm0, gm0, qm0,
                                 wm1, bm1, gm1, qm1, wo, bo, out);
}
